// QORAttention_11476152615185
// MI455X (gfx1250) — matrix operand order repaired and checked by reading, not compiled, not run on hardware
//
#include <hip/hip_runtime.h>
#include <hip/hip_bf16.h>
#include <stdint.h>

#define B_   2
#define T_   2048
#define C_   2048
#define NH   16
#define HD   128
#define NKV  4
#define WINDOW_ 512
#define QKDIM (NH*HD)    // 2048
#define KVDIM (NKV*HD)   // 512
#define KCHUNK 256

typedef __attribute__((ext_vector_type(8)))  float  v8f;
typedef __attribute__((ext_vector_type(8)))  __bf16 v8bf;
typedef __attribute__((ext_vector_type(16))) __bf16 v16bf;

union V16 { v16bf v; v8bf h[2]; };

__device__ __forceinline__ unsigned short f2bf(float f) {
  union { float f; uint32_t u; } c; c.f = f;
  uint32_t u = c.u;
  u += 0x7FFFu + ((u >> 16) & 1u);   // round-to-nearest-even
  return (unsigned short)(u >> 16);
}
__device__ __forceinline__ v8bf ldv8(const unsigned short* p) {
  return *reinterpret_cast<const v8bf*>(p);
}
__device__ __forceinline__ v16bf ldv16(const unsigned short* p) {
  return *reinterpret_cast<const v16bf*>(p);
}

// ---------------- RMSNorm: one block per (b,t) row ----------------
__global__ __launch_bounds__(256)
void rmsnorm_kernel(const float* __restrict__ x, const float* __restrict__ w,
                    unsigned short* __restrict__ out) {
  int row = blockIdx.x;
  int tid = threadIdx.x;
  const float* xr = x + (size_t)row * C_;
  float ss = 0.f;
  #pragma unroll
  for (int i = 0; i < C_/256; ++i) { float v = xr[tid + i*256]; ss += v*v; }
  #pragma unroll
  for (int o = 16; o >= 1; o >>= 1) ss += __shfl_xor(ss, o, 32);
  __shared__ float red[8];
  if ((tid & 31) == 0) red[tid >> 5] = ss;
  __syncthreads();
  float tot = 0.f;
  #pragma unroll
  for (int i = 0; i < 8; ++i) tot += red[i];
  float rms = rsqrtf(tot * (1.0f / C_) + 1e-6f);
  unsigned short* orow = out + (size_t)row * C_;
  #pragma unroll
  for (int i = 0; i < C_/256; ++i) {
    int c = tid + i*256;
    orow[c] = f2bf(xr[c] * rms * w[c]);
  }
}

// ------------- convert fp32 [K][N] -> bf16 transposed [N][K] -------------
__global__ __launch_bounds__(256)
void convt_kernel(const float* __restrict__ w, unsigned short* __restrict__ wt,
                  int K, int N) {
  int idx = blockIdx.x * blockDim.x + threadIdx.x;
  if (idx >= K * N) return;
  int k = idx / N, n = idx - k * N;
  wt[(size_t)n * K + k] = f2bf(w[idx]);
}

// ---------------- bf16 WMMA GEMM ----------------
// C[M][N] = A[M][K] * Bt[N][K]^T (+resid). Block = 8 waves, tile 16(M) x 512(N).
// A panel (16 x KCHUNK) staged once per block in LDS; each wave owns 16x64
// (4 accumulators) so each A fragment feeds 4 WMMAs.
// Requires: M%16==0, N%512==0, K%KCHUNK==0.
__global__ __launch_bounds__(256)
void gemm_bf16_kernel(const unsigned short* __restrict__ A,
                      const unsigned short* __restrict__ Bt,
                      float* __restrict__ C,
                      const float* __restrict__ resid,
                      int M, int N, int K) {
  __shared__ __align__(32) unsigned short As[16][KCHUNK];   // 8 KiB
  int tid  = threadIdx.x;
  int lane = tid & 31;
  int wave = tid >> 5;
  int m0 = blockIdx.x * 16;
  int n0 = blockIdx.y * 512 + wave * 64;
  int nlo   = lane & 15;
  int kb8   = (lane >> 4) * 8;         // unused; A frag also uses khi16
  int khi16 = (lane >> 4) * 16;        // B-frag K half

  const unsigned short* brow[4];
  #pragma unroll
  for (int j = 0; j < 4; ++j) brow[j] = Bt + (size_t)(n0 + j*16 + nlo) * K;

  v8f zero = {};
  v8f acc[4];
  #pragma unroll
  for (int j = 0; j < 4; ++j) acc[j] = zero;

  for (int k0 = 0; k0 < K; k0 += KCHUNK) {
    // cooperative A panel load: 512 x 16B chunks over 256 threads
    #pragma unroll
    for (int t = 0; t < 2; ++t) {
      int c = tid + t*256;
      int r = c >> 5;
      int col = (c & 31) << 3;
      *reinterpret_cast<v8bf*>(&As[r][col]) =
          ldv8(A + (size_t)(m0 + r) * K + k0 + col);
      if (k0 + KCHUNK < K)   // prefetch next A panel (global_prefetch_b8)
        __builtin_prefetch(A + (size_t)(m0 + r) * K + k0 + KCHUNK + col, 0, 3);
    }
    __syncthreads();
    #pragma unroll
    for (int kk = 0; kk < KCHUNK/32; ++kk) {
      V16 a;
      a.h[0] = ldv8(&As[nlo][kk*32 + khi16]);
      a.h[1] = ldv8(&As[nlo][kk*32 + khi16 + 8]);
      #pragma unroll
      for (int j = 0; j < 4; ++j) {
        V16 b; b.v = ldv16(brow[j] + k0 + kk*32 + khi16);
        acc[j] = __builtin_amdgcn_wmma_f32_16x16x32_bf16(false, a.v, false, b.v,
                                                         (short)0, acc[j], false, false);
      }
    }
    __syncthreads();
  }

  int mrow = (lane >> 4) * 8;
  #pragma unroll
  for (int j = 0; j < 4; ++j) {
    #pragma unroll
    for (int i = 0; i < 8; ++i) {
      size_t off = (size_t)(m0 + mrow + i) * N + (n0 + j*16 + nlo);
      float v = acc[j][i];
      if (resid) v += resid[off];
      C[off] = v;
    }
  }
}

// ---------------- RoPE + relayout ----------------
__global__ __launch_bounds__(256)
void rope_q_kernel(const float* __restrict__ qf, const float* __restrict__ cosT,
                   const float* __restrict__ sinT, unsigned short* __restrict__ qb) {
  int idx = blockIdx.x * blockDim.x + threadIdx.x;   // B*T*NH*64
  int i = idx & 63;
  int h = (idx >> 6) & (NH - 1);
  int t = (idx >> 10) & (T_ - 1);
  int b = idx >> 21;
  const float* src = qf + (size_t)(b * T_ + t) * QKDIM + h * HD;
  float x1 = src[i], x2 = src[i + 64];
  float c = cosT[t * 64 + i], s = sinT[t * 64 + i];
  unsigned short* dst = qb + ((size_t)(b * NH + h) * T_ + t) * HD;
  dst[i]      = f2bf(x1 * c - x2 * s);
  dst[i + 64] = f2bf(x2 * c + x1 * s);
}

__global__ __launch_bounds__(256)
void rope_k_kernel(const float* __restrict__ kf, const float* __restrict__ cosT,
                   const float* __restrict__ sinT, unsigned short* __restrict__ kb) {
  int idx = blockIdx.x * blockDim.x + threadIdx.x;   // B*T*NKV*64
  int i   = idx & 63;
  int kvh = (idx >> 6) & (NKV - 1);
  int t   = (idx >> 8) & (T_ - 1);
  int b   = idx >> 19;
  const float* src = kf + (size_t)(b * T_ + t) * KVDIM + kvh * HD;
  float x1 = src[i], x2 = src[i + 64];
  float c = cosT[t * 64 + i], s = sinT[t * 64 + i];
  unsigned short* dst = kb + ((size_t)(b * NKV + kvh) * T_ + t) * HD;
  dst[i]      = f2bf(x1 * c - x2 * s);
  dst[i + 64] = f2bf(x2 * c + x1 * s);
}

__global__ __launch_bounds__(256)
void pack_v_kernel(const float* __restrict__ vf, unsigned short* __restrict__ vt) {
  int idx = blockIdx.x * blockDim.x + threadIdx.x;   // B*T*KVDIM
  int d   = idx & (HD - 1);
  int kvh = (idx >> 7) & (NKV - 1);
  int t   = (idx >> 9) & (T_ - 1);
  int b   = idx >> 20;
  vt[((size_t)(b * NKV + kvh) * HD + d) * T_ + t] = f2bf(vf[idx]);
}

// ---------------- Flash attention (sliding window 512, causal) ----------------
// 4 waves per block; each wave independently owns one (b, h, 16-query) tile.
// No cross-wave sync needed: LDS staging buffers are per-wave, and LDS ops are
// in-order within a wave (wave_barrier only blocks compile-time reordering).
__global__ __launch_bounds__(128)
void attn_kernel(const unsigned short* __restrict__ qb,   // [B][NH][T][HD]
                 const unsigned short* __restrict__ kb,   // [B][NKV][T][HD]
                 const unsigned short* __restrict__ vtb,  // [B][NKV][HD][T]
                 unsigned short* __restrict__ ob)         // [B][T][NH*HD]
{
  const int lane  = threadIdx.x & 31;
  const int wave  = threadIdx.x >> 5;
  const int nlo   = lane & 15;
  const int kb8   = (lane >> 4) * 8;
  const int khi16 = (lane >> 4) * 16;
  const int mrow  = (lane >> 4) * 8;

  int gid = blockIdx.x * 4 + wave;      // global q-tile id over B*NH*(T/16)
  int qt  = gid & (T_/16 - 1);
  int h   = (gid >> 7) & (NH - 1);
  int b   = gid >> 11;
  int kvh = h >> 2;
  int t0  = qt * 16;

  const unsigned short* qbase = qb  + ((size_t)(b*NH  + h  ) * T_) * HD;
  const unsigned short* kbase = kb  + ((size_t)(b*NKV + kvh) * T_) * HD;
  const unsigned short* vbase = vtb + ((size_t)(b*NKV + kvh) * HD) * T_;

  // Q fragments: rows t0..t0+15, K=HD in 4 steps of 32
  V16 qfrag[4];
  {
    const unsigned short* qrow = qbase + (size_t)(t0 + nlo) * HD;
    #pragma unroll
    for (int kk = 0; kk < 4; ++kk) {
      qfrag[kk].h[0] = ldv8(qrow + kk*32 + khi16);
      qfrag[kk].h[1] = ldv8(qrow + kk*32 + khi16 + 8);
    }
  }

  v8f zero = {};
  v8f acc[8];
  #pragma unroll
  for (int d = 0; d < 8; ++d) acc[d] = zero;
  float m_run[8], l_run[8];
  #pragma unroll
  for (int j = 0; j < 8; ++j) { m_run[j] = -3.0e38f; l_run[j] = 0.f; }

  __shared__ __align__(32) unsigned short pt[4][16][32];   // per-wave P staging
  unsigned short (*ptw)[32] = pt[wave];
  const float scale = 0.08838834764831845f;   // 1/sqrt(128)

  int sLo = t0 - (WINDOW_ - 1); if (sLo < 0) sLo = 0; sLo &= ~31;
  int sHi = t0 + 15;

  for (int sb = sLo; sb <= sHi; sb += 32) {
    // ---- scores: Q (16xHD) x K^T (HDx32), two 16-key n-tiles ----
    v8f sc[2];
    #pragma unroll
    for (int nt = 0; nt < 2; ++nt) {
      int scol = sb + nt*16 + nlo;              // key index = this lane's B column
      const unsigned short* krow = kbase + (size_t)scol * HD;
      v8f s8 = zero;
      #pragma unroll
      for (int kk = 0; kk < 4; ++kk) {
        V16 bb; bb.v = ldv16(krow + kk*32 + khi16);
        s8 = __builtin_amdgcn_wmma_f32_16x16x32_bf16(false, qfrag[kk].v, false, bb.v,
                                                     (short)0, s8, false, false);
      }
      sc[nt] = s8;
    }
    // ---- mask + online softmax (row = t0+mrow+j, cols across 16 lanes) ----
    float fac[8];
    #pragma unroll
    for (int j = 0; j < 8; ++j) {
      int t = t0 + mrow + j;
      #pragma unroll
      for (int nt = 0; nt < 2; ++nt) {
        int s = sb + nt*16 + nlo;
        float v = sc[nt][j] * scale;
        bool ok = (s <= t) && ((t - s) < WINDOW_);
        sc[nt][j] = ok ? v : -1e30f;
      }
      float mv = fmaxf(sc[0][j], sc[1][j]);
      #pragma unroll
      for (int o = 8; o >= 1; o >>= 1) mv = fmaxf(mv, __shfl_xor(mv, o, 16));
      float mnew = fmaxf(m_run[j], mv);
      float f  = __expf(m_run[j] - mnew);
      float p0 = __expf(sc[0][j] - mnew);
      float p1 = __expf(sc[1][j] - mnew);
      sc[0][j] = p0; sc[1][j] = p1;
      float ps = p0 + p1;
      #pragma unroll
      for (int o = 8; o >= 1; o >>= 1) ps += __shfl_xor(ps, o, 16);
      l_run[j] = l_run[j] * f + ps;
      m_run[j] = mnew;
      fac[j] = f;
    }
    #pragma unroll
    for (int d = 0; d < 8; ++d)
      #pragma unroll
      for (int j = 0; j < 8; ++j) acc[d][j] *= fac[j];

    // ---- stage P (16x32) to per-wave LDS, reload as WMMA A-fragment ----
    #pragma unroll
    for (int nt = 0; nt < 2; ++nt)
      #pragma unroll
      for (int j = 0; j < 8; ++j)
        ptw[mrow + j][nt*16 + nlo] = f2bf(sc[nt][j]);
    __builtin_amdgcn_wave_barrier();    // LDS in-order per wave; block reordering
    V16 pfrag;
    pfrag.h[0] = ldv8(&ptw[nlo][0] + khi16);
    pfrag.h[1] = ldv8(&ptw[nlo][0] + khi16 + 8);
    // ---- acc += P (16x32) x V (32x16 per d-tile), V stored transposed [D][T] ----
    #pragma unroll
    for (int d = 0; d < 8; ++d) {
      int dcol = d*16 + nlo;
      const unsigned short* vrow = vbase + (size_t)dcol * T_;
      V16 vb; vb.v = ldv16(vrow + sb + khi16);
      acc[d] = __builtin_amdgcn_wmma_f32_16x16x32_bf16(false, pfrag.v, false, vb.v,
                                                       (short)0, acc[d], false, false);
    }
    __builtin_amdgcn_wave_barrier();
  }

  float inv_l[8];
  #pragma unroll
  for (int j = 0; j < 8; ++j) inv_l[j] = (l_run[j] > 0.f) ? (1.f / l_run[j]) : 0.f;
  #pragma unroll
  for (int d = 0; d < 8; ++d)
    #pragma unroll
    for (int j = 0; j < 8; ++j) {
      int t = t0 + mrow + j;
      size_t off = ((size_t)b * T_ + t) * QKDIM + h*HD + d*16 + nlo;
      ob[off] = f2bf(acc[d][j] * inv_l[j]);
    }
}

// ---------------- host-side orchestration ----------------
extern "C" void kernel_launch(void* const* d_in, const int* in_sizes, int n_in,
                              void* d_out, int out_size, void* d_ws, size_t ws_size,
                              hipStream_t stream) {
  const float* x  = (const float*)d_in[0];
  const float* nw = (const float*)d_in[1];
  const float* wq = (const float*)d_in[2];
  const float* wk = (const float*)d_in[3];
  const float* wv = (const float*)d_in[4];
  const float* wo = (const float*)d_in[5];
  const float* rc = (const float*)d_in[6];
  const float* rs = (const float*)d_in[7];
  float* out = (float*)d_out;
  char* ws = (char*)d_ws;

  // workspace layout (bytes)
  unsigned short* normed = (unsigned short*)(ws + 0);          // 16 MiB
  unsigned short* wqT    = (unsigned short*)(ws + 16777216);   //  8 MiB  [2048][2048]
  unsigned short* wkT    = (unsigned short*)(ws + 25165824);   //  2 MiB  [512][2048]
  unsigned short* wvT    = (unsigned short*)(ws + 27262976);   //  2 MiB
  unsigned short* woT    = (unsigned short*)(ws + 29360128);   //  8 MiB
  float*          qf     = (float*)(ws + 37748736);            // 32 MiB
  float*          kf     = (float*)(ws + 71303168);            //  8 MiB
  float*          vf     = (float*)(ws + 79691776);            //  8 MiB
  unsigned short* qbp    = (unsigned short*)(ws + 88080384);   // 16 MiB [B][NH][T][HD]
  unsigned short* kbp    = (unsigned short*)(ws + 104857600);  //  4 MiB [B][NKV][T][HD]
  unsigned short* vtp    = (unsigned short*)(ws + 109051904);  //  4 MiB [B][NKV][HD][T]
  unsigned short* attnb  = (unsigned short*)(ws + 37748736);   // aliases qf (dead by then)

  const int MT = B_ * T_;        // 4096 rows

  rmsnorm_kernel<<<MT, 256, 0, stream>>>(x, nw, normed);

  convt_kernel<<<(C_*QKDIM + 255)/256, 256, 0, stream>>>(wq, wqT, C_, QKDIM);
  convt_kernel<<<(C_*KVDIM + 255)/256, 256, 0, stream>>>(wk, wkT, C_, KVDIM);
  convt_kernel<<<(C_*KVDIM + 255)/256, 256, 0, stream>>>(wv, wvT, C_, KVDIM);
  convt_kernel<<<(QKDIM*C_ + 255)/256, 256, 0, stream>>>(wo, woT, QKDIM, C_);

  dim3 gq(MT/16, QKDIM/512);
  gemm_bf16_kernel<<<gq, 256, 0, stream>>>(normed, wqT, qf, nullptr, MT, QKDIM, C_);
  dim3 gkv(MT/16, KVDIM/512);
  gemm_bf16_kernel<<<gkv, 256, 0, stream>>>(normed, wkT, kf, nullptr, MT, KVDIM, C_);
  gemm_bf16_kernel<<<gkv, 256, 0, stream>>>(normed, wvT, vf, nullptr, MT, KVDIM, C_);

  rope_q_kernel<<<(B_*T_*NH*64)/256, 256, 0, stream>>>(qf, rc, rs, qbp);
  rope_k_kernel<<<(B_*T_*NKV*64)/256, 256, 0, stream>>>(kf, rc, rs, kbp);
  pack_v_kernel<<<(B_*T_*KVDIM)/256, 256, 0, stream>>>(vf, vtp);

  attn_kernel<<<B_*NH*(T_/16)/4, 128, 0, stream>>>(qbp, kbp, vtp, attnb);

  dim3 go(MT/16, C_/512);
  gemm_bf16_kernel<<<go, 256, 0, stream>>>(attnb, woT, out, x, MT, C_, C_);
}
